// CatConLayers_33638183863042
// MI455X (gfx1250) — matrix operand order repaired and checked by reading, not compiled, not run on hardware
//
#include <hip/hip_runtime.h>

typedef __attribute__((ext_vector_type(16))) _Float16 v16h;
typedef __attribute__((ext_vector_type(8)))  _Float16 v8h;
typedef __attribute__((ext_vector_type(8)))  float    v8f;

#define NB   4
#define TT   1024
#define HH   4
#define KQD  128
#define LDV  128
#define NRF  128
#define HKW  512   // H*KQ

__device__ __forceinline__ v8f wmma16(v16h a, v16h b, v8f c) {
  return __builtin_amdgcn_wmma_f32_16x16x32_f16(false, a, false, b, (short)0, c, false, false);
}

// ---------------------------------------------------------------------------
// Kernel 1: build k_in [4096,128] f16, q_in [128,128] f16, xT [4,128,1024] f16
// ---------------------------------------------------------------------------
__global__ __launch_bounds__(128)
void prep_kernel(const float* __restrict__ ts, const int* __restrict__ ys0,
                 const int* __restrict__ ys1, const float* __restrict__ x,
                 const float* __restrict__ emb0, const float* __restrict__ emb1,
                 _Float16* __restrict__ k_in, _Float16* __restrict__ q_in,
                 _Float16* __restrict__ xT) {
  const int NKROWS = NB * TT;            // 4096
  int b = blockIdx.x;
  int tid = threadIdx.x;                 // 128 threads
  if (b < NKROWS + NRF) {
    bool is_q = (b >= NKROWS);
    int row = is_q ? (b - NKROWS) : b;
    float t  = is_q ? ((float)row * (1.0f / (NRF - 1))) : ts[row];
    int y0 = is_q ? 100 : ys0[row];
    int y1 = is_q ? 50  : ys1[row];
    int col = tid;
    float val;
    if (col < 64) {
      int i = col >> 1;
      float div = __expf((float)(2 * i) * (-2.302585092994046f / 64.0f));
      float ang = 48.0f * t * div;
      val = (col & 1) ? __cosf(ang) : __sinf(ang);
    } else if (col < 96) {
      val = emb0[y0 * 32 + (col - 64)];
    } else {
      val = emb1[y1 * 32 + (col - 96)];
    }
    _Float16* dst = is_q ? (q_in + (size_t)row * KQD) : (k_in + (size_t)row * KQD);
    dst[col] = (_Float16)val;
  } else {
    int r = b - (NKROWS + NRF);          // 0..511 over (n,d)
    int n = r >> 7, d = r & 127;
    for (int tt = tid; tt < TT; tt += 128)
      xT[((size_t)(n * 128 + d)) * TT + tt] = (_Float16)x[((size_t)(n * TT + tt)) * LDV + d];
  }
}

// ---------------------------------------------------------------------------
// Kernel 1b: one-time weight staging:  dst[N,K] f16 = transpose(src[K,N] f32)
// grid = N blocks, 128 threads loop over K.
// ---------------------------------------------------------------------------
__global__ __launch_bounds__(128)
void transpose_w(const float* __restrict__ src, _Float16* __restrict__ dst,
                 int K, int N) {
  int n = blockIdx.x;
  for (int k = threadIdx.x; k < K; k += 128)
    dst[(size_t)n * K + k] = (_Float16)src[(size_t)k * N + n];
}

// ---------------------------------------------------------------------------
// Kernel 2: C[M,N] = A[M,K](f16) @ BT[N,K](f16, pre-transposed) + bias.
// Wave = 16x32 tile. grid = (M/128, N/32), block = 256 (8 waves along M).
// All fragment loads are contiguous 16/32-byte vector loads.
// ---------------------------------------------------------------------------
__global__ __launch_bounds__(256)
void gemm_kernel(const _Float16* __restrict__ A, const _Float16* __restrict__ BT,
                 const float* __restrict__ bias, _Float16* __restrict__ outH,
                 float* __restrict__ outF, int M, int N, int K) {
  int wave = threadIdx.x >> 5;
  int lane = threadIdx.x & 31;
  int grp  = lane >> 4;                  // half-wave group
  int colN = lane & 15;                  // B/C column within tile
  int rowA = lane & 15;                  // A row within tile
  int m0 = blockIdx.x * 128 + wave * 16;
  int n0 = blockIdx.y * 32;

  v8f acc0, acc1;
  float bv0 = bias[n0 + colN];
  float bv1 = bias[n0 + 16 + colN];
#pragma unroll
  for (int v = 0; v < 8; v++) { acc0[v] = bv0; acc1[v] = bv1; }

  const _Float16* ap0 = A + (size_t)(m0 + rowA) * K + grp * 16;
  const _Float16* bp0 = BT + (size_t)(n0 + colN) * K + grp * 16;
  const _Float16* bp1 = BT + (size_t)(n0 + 16 + colN) * K + grp * 16;

  for (int k0 = 0; k0 < K; k0 += 32) {
    // A fragment: this kernel's k order (M=lane&15, K = 16*grp + j), contiguous, same as B
    v8h alo = *(const v8h*)(ap0 + k0);
    v8h ahi = *(const v8h*)(ap0 + k0 + 8);
    v16h af;
#pragma unroll
    for (int j = 0; j < 8; j++) { af[j] = alo[j]; af[j + 8] = ahi[j]; }
    // B fragments: same k order as A (col=lane&15, K = 16*grp + j), contiguous
    v16h bf0 = *(const v16h*)(bp0 + k0);
    v16h bf1 = *(const v16h*)(bp1 + k0);
    acc0 = wmma16(af, bf0, acc0);
    acc1 = wmma16(af, bf1, acc1);
  }
  // C/D layout: N = lane&15, M = v + 8*grp
#pragma unroll
  for (int v = 0; v < 8; v++) {
    int mr = m0 + v + 8 * grp;
    if (outH) {
      outH[(size_t)mr * N + n0 + colN]      = (_Float16)acc0[v];
      outH[(size_t)mr * N + n0 + 16 + colN] = (_Float16)acc1[v];
    } else {
      outF[(size_t)mr * N + n0 + colN]      = acc0[v];
      outF[(size_t)mr * N + n0 + 16 + colN] = acc1[v];
    }
  }
}

// ---------------------------------------------------------------------------
// Kernel 3: fused attention per (n,h).  Block = 8 waves x 16 query rows.
// Flash-style online softmax over T=1024 keys in chunks of 32.
// ---------------------------------------------------------------------------
__global__ __launch_bounds__(256)
void attn_kernel(const _Float16* __restrict__ hq, const _Float16* __restrict__ hk,
                 const _Float16* __restrict__ xT, _Float16* __restrict__ att) {
  __shared__ _Float16 ldsP[8 * 16 * 32];          // per-wave 16x32 P tile
  int n = blockIdx.x >> 2;
  int h = blockIdx.x & 3;
  int wave = threadIdx.x >> 5;
  int lane = threadIdx.x & 31;
  int grp  = lane >> 4;
  int colN = lane & 15;
  int q0   = wave * 16;

  // Q A-fragments for this wave's 16 rows, 4 chunks of K=32 over d=0..127
  v16h qf[4];
  {
    const _Float16* qp = hq + (size_t)(q0 + colN) * HKW + h * KQD;
#pragma unroll
    for (int c = 0; c < 4; c++) {
      const _Float16* p = qp + c * 32 + grp * 16;
      v8h lo = *(const v8h*)p;
      v8h hi = *(const v8h*)(p + 8);
#pragma unroll
      for (int j = 0; j < 8; j++) { qf[c][j] = lo[j]; qf[c][j + 8] = hi[j]; }
    }
  }

  v8f O[8];
#pragma unroll
  for (int d = 0; d < 8; d++)
#pragma unroll
    for (int v = 0; v < 8; v++) O[d][v] = 0.0f;
  float m[8], l[8];
#pragma unroll
  for (int v = 0; v < 8; v++) { m[v] = -1e30f; l[v] = 0.0f; }

  const float sc = 0.0883883476483184f;           // 1/sqrt(128)

  for (int t0 = 0; t0 < TT; t0 += 32) {
    // prefetch next chunk's K rows and V columns (global_prefetch_b8)
    if (t0 + 32 < TT) {
      __builtin_prefetch(hk + (size_t)(n * TT + t0 + 32 + colN) * HKW + h * KQD, 0, 0);
      __builtin_prefetch(xT + (size_t)(n * 128 + grp * 64 + colN) * TT + t0 + 32, 0, 0);
    }
    // --- scores: two 16x16 tiles covering 32 keys ---
    v8f S[2];
#pragma unroll
    for (int half = 0; half < 2; half++) {
      v8f s;
#pragma unroll
      for (int v = 0; v < 8; v++) s[v] = 0.0f;
      int tb = t0 + half * 16;
      const _Float16* kp = hk + (size_t)(n * TT + tb + colN) * HKW + h * KQD + grp * 16;
#pragma unroll
      for (int c = 0; c < 4; c++) {
        v16h bf = *(const v16h*)(kp + c * 32);
        s = wmma16(qf[c], bf, s);
      }
      S[half] = s;
    }
    // --- online softmax statistics (row = v + 8*grp, spread over 16 lanes) ---
    float mn[8], corr[8];
#pragma unroll
    for (int v = 0; v < 8; v++) {
      S[0][v] *= sc; S[1][v] *= sc;
      float rm = fmaxf(S[0][v], S[1][v]);
#pragma unroll
      for (int off = 1; off < 16; off <<= 1) rm = fmaxf(rm, __shfl_xor(rm, off, 32));
      mn[v] = fmaxf(m[v], rm);
      corr[v] = __expf(m[v] - mn[v]);
      m[v] = mn[v];
    }
#pragma unroll
    for (int v = 0; v < 8; v++) {
      S[0][v] = __expf(S[0][v] - mn[v]);
      S[1][v] = __expf(S[1][v] - mn[v]);
      float rs = S[0][v] + S[1][v];
#pragma unroll
      for (int off = 1; off < 16; off <<= 1) rs += __shfl_xor(rs, off, 32);
      l[v] = l[v] * corr[v] + rs;
#pragma unroll
      for (int d = 0; d < 8; d++) O[d][v] *= corr[v];
    }
    // --- P (C/D layout) -> LDS -> A-fragment layout (wave-private, in-order DS) ---
    _Float16* lp = ldsP + wave * 512;
#pragma unroll
    for (int half = 0; half < 2; half++)
#pragma unroll
      for (int v = 0; v < 8; v++)
        lp[(v + 8 * grp) * 32 + half * 16 + colN] = (_Float16)S[half][v];
    v16h pf;
    {
      const _Float16* pp = lp + colN * 32 + grp * 16;
      v8h lo = *(const v8h*)pp;
      v8h hi = *(const v8h*)(pp + 8);
#pragma unroll
      for (int j = 0; j < 8; j++) { pf[j] = lo[j]; pf[j + 8] = hi[j]; }
    }
    // --- O += P @ V  (V from xT, contiguous in key dim) ---
#pragma unroll
    for (int d = 0; d < 8; d++) {
      const _Float16* vp = xT + (size_t)(n * 128 + d * 16 + colN) * TT + t0 + grp * 16;
      v16h vf = *(const v16h*)vp;
      O[d] = wmma16(pf, vf, O[d]);
    }
  }

  // normalize and store: att[n, q, h*128 + vd]
#pragma unroll
  for (int d = 0; d < 8; d++) {
#pragma unroll
    for (int v = 0; v < 8; v++) {
      int mr = q0 + v + 8 * grp;
      float inv = 1.0f / l[v];
      att[(size_t)(n * NRF + mr) * HKW + h * LDV + d * 16 + colN] = (_Float16)(O[d][v] * inv);
    }
  }
}

// ---------------------------------------------------------------------------
extern "C" void kernel_launch(void* const* d_in, const int* in_sizes, int n_in,
                              void* d_out, int out_size, void* d_ws, size_t ws_size,
                              hipStream_t stream) {
  const float* ts   = (const float*)d_in[0];
  const int*   ys0  = (const int*)d_in[1];
  const int*   ys1  = (const int*)d_in[2];
  const float* x    = (const float*)d_in[3];
  const float* emb0 = (const float*)d_in[4];
  const float* emb1 = (const float*)d_in[5];
  const float* Wq   = (const float*)d_in[6];
  const float* bq   = (const float*)d_in[7];
  const float* Wk   = (const float*)d_in[8];
  const float* bk   = (const float*)d_in[9];
  const float* Wo   = (const float*)d_in[10];
  const float* bo   = (const float*)d_in[11];

  char* ws = (char*)d_ws;
  _Float16* k_in = (_Float16*)ws; ws += (size_t)4096 * 128 * 2;
  _Float16* q_in = (_Float16*)ws; ws += (size_t)128 * 128 * 2;
  _Float16* xT   = (_Float16*)ws; ws += (size_t)4 * 128 * 1024 * 2;
  _Float16* hk   = (_Float16*)ws; ws += (size_t)4096 * 512 * 2;
  _Float16* hq   = (_Float16*)ws; ws += (size_t)128 * 512 * 2;
  _Float16* att  = (_Float16*)ws; ws += (size_t)512 * 512 * 2;
  _Float16* WkT  = (_Float16*)ws; ws += (size_t)512 * 128 * 2;
  _Float16* WqT  = (_Float16*)ws; ws += (size_t)512 * 128 * 2;
  _Float16* WoT  = (_Float16*)ws; ws += (size_t)128 * 512 * 2;

  // 1) embeddings + x transpose + weight staging (f32 -> transposed f16)
  prep_kernel<<<4736, 128, 0, stream>>>(ts, ys0, ys1, x, emb0, emb1, k_in, q_in, xT);
  transpose_w<<<512, 128, 0, stream>>>(Wk, WkT, 128, 512);
  transpose_w<<<512, 128, 0, stream>>>(Wq, WqT, 128, 512);
  transpose_w<<<128, 128, 0, stream>>>(Wo, WoT, 512, 128);
  // 2) key / query projections (WMMA GEMMs)
  gemm_kernel<<<dim3(32, 16), 256, 0, stream>>>(k_in, WkT, bk, hk, nullptr, 4096, 512, 128);
  gemm_kernel<<<dim3(1, 16),  256, 0, stream>>>(q_in, WqT, bq, hq, nullptr, 128, 512, 128);
  // 3) fused flash-attention per (n,h)
  attn_kernel<<<16, 256, 0, stream>>>(hq, hk, xT, att);
  // 4) output projection -> d_out (f32)
  gemm_kernel<<<dim3(4, 4), 256, 0, stream>>>(att, WoT, bo, nullptr, (float*)d_out, 512, 128, 512);
}
